// CoaT_9105330668168
// MI455X (gfx1250) — matrix operand order repaired and checked by reading, not compiled, not run on hardware
//
#include <hip/hip_runtime.h>

// ---------------------------------------------------------------------------
// CoaT factorized-attention block for MI455X (gfx1250, wave32, WMMA).
//   1) qkv = x @ Wqkv              f16-WMMA GEMM, async-LDS double buffered
//   2) softmax denominators of k   sum(exp) w/ f32 atomics
//   3) kv = softmax(k)^T @ v       LDS outer products, 40x40/head
//   4) depthwise conv 3/5/7 on v   8-channel groups, LDS halo tiles
//   5) attn = scale*q@kv + crpe    -> f16 intermediate
//   6) out = attn @ Wproj + bproj  f16-WMMA GEMM (f16 A path)
// GEMM staging uses global_load_async_to_lds_b128 (ASYNCcnt) so tile k+1
// streams into LDS while tile k is consumed by v_wmma_f32_16x16x32_f16.
// ---------------------------------------------------------------------------

typedef _Float16 h8   __attribute__((ext_vector_type(8)));
typedef _Float16 h16  __attribute__((ext_vector_type(16)));
typedef float    f32x4 __attribute__((ext_vector_type(4)));
typedef float    f32x8 __attribute__((ext_vector_type(8)));

namespace coat {
constexpr int B    = 8;
constexpr int N    = 12545;       // 1 + 112*112
constexpr int C    = 320;
constexpr int NH   = 8;
constexpr int CH   = 40;
constexpr int HH   = 112;
constexpr int WW   = 112;
constexpr int MTOT = B * N;       // 100360
}

// Low 32 bits of a flat shared-memory pointer == LDS byte offset (LDS
// aperture keeps the offset in addr[31:0]); this is what the async-to-LDS
// instruction expects in its per-lane LDS-address VGPR.
__device__ __forceinline__ unsigned lds_addr_of(const void* p) {
  return (unsigned)(unsigned long long)p;
}

__device__ __forceinline__ void async_copy_b128(unsigned lds_off, const void* gaddr) {
  asm volatile("global_load_async_to_lds_b128 %0, %1, off"
               :: "v"(lds_off), "v"(gaddr) : "memory");
}

// ---------------------------------------------------------------------------
// WMMA GEMM: C[M,Nc] = A[M,K] * W[K,Nc] (+bias).  A fp32 or f16, W fp32.
// Block tile 128x64, BK=32, 256 threads = 8 waves (4x2), wave tile 32x32
// (= 2x2 v_wmma_f32_16x16x32_f16).  Double-buffered async-LDS staging.
// ---------------------------------------------------------------------------
template <typename TA, bool HAS_BIAS>
__global__ __launch_bounds__(256) void gemm_wmma_async(
    const TA* __restrict__ A, int lda,
    const float* __restrict__ W, int ldw,
    float* __restrict__ Cout, int ldc,
    const float* __restrict__ bias,
    int M, int Nc, int K)
{
  constexpr int BM = 128, BN = 64, BK = 32;
  constexpr bool A_IS_F32 = (sizeof(TA) == 4);
  constexpr int APITCH = A_IS_F32 ? 144 : 80;       // bytes/row in LDS (16B mult)
  constexpr int BPITCH = 272;                       // 64 floats + pad
  constexpr int ACH_ROW = (BK * (int)sizeof(TA)) / 16; // 16B chunks per A row: 8|4
  constexpr int AITER = (BM * ACH_ROW) / 256;          // async ops/thread for A: 4|2

  __shared__ __align__(16) char sA[2][BM * APITCH];
  __shared__ __align__(16) char sB[2][BK * BPITCH];

  const int tid  = threadIdx.x;
  const int wave = tid >> 5;
  const int lane = tid & 31;
  const int wm = (wave >> 1) * 32;   // wave row in block tile
  const int wn = (wave & 1) * 32;    // wave col in block tile

  const int blockM = blockIdx.x * BM;
  const int blockN = blockIdx.y * BN;

  // Issue one K-tile's worth of async copies.  Exec-uniform: every thread
  // always issues AITER+2 ops so per-wave ASYNCcnt bookkeeping is exact;
  // M-edge rows are clamped (their garbage lands in rows we never store).
  auto issue_tile = [&](int buf, int k0) {
    #pragma unroll
    for (int i = 0; i < AITER; ++i) {
      const int cid  = tid + i * 256;
      const int row  = cid / ACH_ROW;
      const int coff = (cid % ACH_ROW) * 16;
      int gr = blockM + row; if (gr > M - 1) gr = M - 1;
      const char* g = (const char*)A + ((size_t)gr * lda + k0) * sizeof(TA) + coff;
      async_copy_b128(lds_addr_of(&sA[buf][row * APITCH + coff]), g);
    }
    #pragma unroll
    for (int i = 0; i < 2; ++i) {
      const int cid  = tid + i * 256;
      const int row  = cid >> 4;           // 16 chunks per 64-float row
      const int coff = (cid & 15) * 16;
      const char* g = (const char*)W + ((size_t)(k0 + row) * ldw + blockN) * 4 + coff;
      async_copy_b128(lds_addr_of(&sB[buf][row * BPITCH + coff]), g);
    }
  };

  f32x8 acc[2][2] = {};

  issue_tile(0, 0);
  const int NT = K / BK;                   // K is a multiple of 32
  for (int it = 0; it < NT; ++it) {
    const int cur = it & 1;
    __syncthreads();                       // everyone done reading buf cur^1
    if (it + 1 < NT) {
      issue_tile(cur ^ 1, (it + 1) * BK);
      // wait until only the just-issued ops remain -> buffer `cur` has landed
      if constexpr (A_IS_F32) asm volatile("s_wait_asynccnt 0x6" ::: "memory");
      else                    asm volatile("s_wait_asynccnt 0x4" ::: "memory");
    } else {
      asm volatile("s_wait_asynccnt 0x0" ::: "memory");
    }
    __syncthreads();                       // all waves' tiles visible

    // ---- A fragments (16-bit wave32 layout: lanes 0-15 row=lane, K 0..7 &
    // 16..23; lanes 16-31 row=lane-16, K 8..15 & 24..31) ----
    h16 afrag[2];
    {
      const int lrow = lane & 15;
      const int kb   = (lane >> 4) * 8;
      #pragma unroll
      for (int tm = 0; tm < 2; ++tm) {
        h16 f;
        if constexpr (A_IS_F32) {
          const float* base = (const float*)&sA[cur][(wm + tm * 16 + lrow) * APITCH] + kb;
          f32x4 a0 = *(const f32x4*)(base);
          f32x4 a1 = *(const f32x4*)(base + 4);
          f32x4 a2 = *(const f32x4*)(base + 16);
          f32x4 a3 = *(const f32x4*)(base + 20);
          #pragma unroll
          for (int i = 0; i < 4; ++i) {
            f[i]      = (_Float16)a0[i];
            f[i + 4]  = (_Float16)a1[i];
            f[i + 8]  = (_Float16)a2[i];
            f[i + 12] = (_Float16)a3[i];
          }
        } else {
          const _Float16* base = (const _Float16*)&sA[cur][(wm + tm * 16 + lrow) * APITCH] + kb;
          h8 lo = *(const h8*)(base);
          h8 hi = *(const h8*)(base + 16);
          #pragma unroll
          for (int i = 0; i < 8; ++i) { f[i] = lo[i]; f[i + 8] = hi[i]; }
        }
        afrag[tm] = f;
      }
    }
    // ---- B fragments (lane L holds column N=L%16; K 0..7 & 16..23 for lanes 0-15, K 8..15 & 24..31 for lanes 16-31) ----
    h16 bfrag[2];
    {
      const int bkb = (lane >> 4) * 8;
      #pragma unroll
      for (int tn = 0; tn < 2; ++tn) {
        const float* base = (const float*)&sB[cur][0] + wn + tn * 16 + (lane & 15);
        h16 f;
        #pragma unroll
        for (int i = 0; i < 8; ++i) {
          f[i]     = (_Float16)base[(bkb + i) * (BPITCH / 4)];
          f[i + 8] = (_Float16)base[(bkb + 16 + i) * (BPITCH / 4)];
        }
        bfrag[tn] = f;
      }
    }

    #pragma unroll
    for (int tm = 0; tm < 2; ++tm)
      #pragma unroll
      for (int tn = 0; tn < 2; ++tn)
        acc[tm][tn] = __builtin_amdgcn_wmma_f32_16x16x32_f16(
            false, afrag[tm], false, bfrag[tn],
            (short)0, acc[tm][tn], false, false);
  }

  // ---- store: C/D layout: VGPR r, lanes 0-15 -> M=r, lanes 16-31 -> M=8+r ----
  const int rbase = (lane >> 4) * 8;
  const int cbase = lane & 15;
  #pragma unroll
  for (int tm = 0; tm < 2; ++tm) {
    #pragma unroll
    for (int tn = 0; tn < 2; ++tn) {
      const int colg = blockN + wn + tn * 16 + cbase;
      const float bv = HAS_BIAS ? bias[colg] : 0.f;
      #pragma unroll
      for (int r = 0; r < 8; ++r) {
        const int rowg = blockM + wm + tm * 16 + rbase + r;
        if (rowg < M)
          Cout[(size_t)rowg * ldc + colg] = acc[tm][tn][r] + bv;
      }
    }
  }
}

// ---------------------------------------------------------------------------
// Softmax denominators: stats[b*320+c] = sum_n exp(k[b,n,c]).
// (|k| is small; the max-free sum is exactly softmax's denominator.)
// ---------------------------------------------------------------------------
__global__ __launch_bounds__(320) void k_sumexp(
    const float* __restrict__ qkv, float* __restrict__ stats)
{
  using namespace coat;
  const int b = blockIdx.y;
  const int c = threadIdx.x;
  const int n0 = blockIdx.x * 393;
  const int n1 = min(N, n0 + 393);
  float s = 0.f;
  for (int n = n0; n < n1; ++n)
    s += __expf(qkv[((size_t)b * N + n) * 960 + C + c]);
  atomicAdd(&stats[b * C + c], s);
}

// ---------------------------------------------------------------------------
// kv[b,h,kk,vv] = sum_n softmax(k)[n,kk] * v[n,vv].  grid (16 chunks, 64 bh).
// ---------------------------------------------------------------------------
__global__ __launch_bounds__(256) void k_kv(
    const float* __restrict__ qkv, const float* __restrict__ stats,
    float* __restrict__ kvb)
{
  using namespace coat;
  __shared__ float sk[64 * CH];
  __shared__ float sv[64 * CH];
  __shared__ float rs[CH];
  const int bh = blockIdx.y;
  const int b = bh >> 3, h = bh & 7;
  const int n0 = blockIdx.x * 785;
  const int n1 = min(N, n0 + 785);
  if (threadIdx.x < CH)
    rs[threadIdx.x] = 1.0f / stats[b * C + h * CH + threadIdx.x];

  float acc[7] = {0.f, 0.f, 0.f, 0.f, 0.f, 0.f, 0.f};

  for (int t0 = n0; t0 < n1; t0 += 64) {
    const int cnt = min(64, n1 - t0);
    __syncthreads();
    for (int e = threadIdx.x; e < cnt * CH; e += 256) {
      const int t = e / CH, cc = e % CH;
      const size_t row = ((size_t)b * N + t0 + t) * 960;
      sk[t * CH + cc] = __expf(qkv[row + C + h * CH + cc]) * rs[cc];
      sv[t * CH + cc] = qkv[row + 2 * C + h * CH + cc];
    }
    __syncthreads();
    #pragma unroll
    for (int i = 0; i < 7; ++i) {
      const int p = threadIdx.x + i * 256;
      if (p < CH * CH) {
        const int kk = p / CH, vv = p % CH;
        float a = acc[i];
        for (int t = 0; t < cnt; ++t)
          a += sk[t * CH + kk] * sv[t * CH + vv];
        acc[i] = a;
      }
    }
  }
  #pragma unroll
  for (int i = 0; i < 7; ++i) {
    const int p = threadIdx.x + i * 256;
    if (p < CH * CH)
      atomicAdd(&kvb[(size_t)bh * CH * CH + p], acc[i]);
  }
}

// ---------------------------------------------------------------------------
// Depthwise conv on v image.  grid (7 y-strips, 40 groups of 8 ch, 8 b).
// Groups of 8 never straddle the 3x3/5x5/7x7 boundaries (80, 200).
// ---------------------------------------------------------------------------
__global__ __launch_bounds__(256) void k_dwconv(
    const float* __restrict__ qkv,
    const float* __restrict__ cw3, const float* __restrict__ cb3,
    const float* __restrict__ cw5, const float* __restrict__ cb5,
    const float* __restrict__ cw7, const float* __restrict__ cb7,
    float* __restrict__ conv_v)
{
  using namespace coat;
  __shared__ float st[8 * 22 * 118];   // 8 ch x (16+2R)x(112+2R), R<=3
  __shared__ float wt[8 * 49];
  const int ystrip = blockIdx.x;
  const int c0 = blockIdx.y * 8;
  const int b = blockIdx.z;

  int R, woff;
  const float *wsrc, *bsrc;
  if (c0 < 80)        { R = 1; wsrc = cw3; bsrc = cb3; woff = c0; }
  else if (c0 < 200)  { R = 2; wsrc = cw5; bsrc = cb5; woff = c0 - 80; }
  else                { R = 3; wsrc = cw7; bsrc = cb7; woff = c0 - 200; }
  const int W2 = 2 * R + 1, WSQ = W2 * W2;

  for (int e = threadIdx.x; e < 8 * WSQ; e += 256) {
    const int cl = e / WSQ, q = e % WSQ;
    wt[cl * 49 + q] = wsrc[(woff + cl) * WSQ + q];
  }
  const int y0 = ystrip * 16;
  const int HS = 16 + 2 * R, WS = WW + 2 * R;
  for (int e = threadIdx.x; e < HS * WS * 8; e += 256) {
    const int cl = e & 7;
    const int xy = e >> 3;
    const int lx = xy % WS, ly = xy / WS;
    const int gy = y0 - R + ly, gx = lx - R;
    float vv = 0.f;
    if (gy >= 0 && gy < HH && gx >= 0 && gx < WW)
      vv = qkv[((size_t)b * N + 1 + gy * WW + gx) * 960 + 2 * C + c0 + cl];
    st[(cl * 22 + ly) * 118 + lx] = vv;
  }
  __syncthreads();

  for (int o = threadIdx.x; o < 16 * WW * 8; o += 256) {
    const int x = o % WW;
    const int y = (o / WW) % 16;
    const int cl = o / (WW * 16);
    float acc = bsrc[woff + cl];
    for (int ky = 0; ky < W2; ++ky)
      for (int kx = 0; kx < W2; ++kx)
        acc += st[(cl * 22 + y + ky) * 118 + x + kx] * wt[cl * 49 + ky * W2 + kx];
    conv_v[(((size_t)b * C + c0 + cl) * HH + y0 + y) * WW + x] = acc;
  }
}

// ---------------------------------------------------------------------------
// attn[b,n,c] = scale * (q[b,n,h*40..] . kv[b,h,:,cv]) + (n>0) q*conv.
// kv for all heads (51KB) + 64 q rows (80KB) live in the 320KB WGP LDS.
// ---------------------------------------------------------------------------
__global__ __launch_bounds__(256) void k_combine(
    const float* __restrict__ qkv, const float* __restrict__ kvb,
    const float* __restrict__ conv_v, _Float16* __restrict__ attn)
{
  using namespace coat;
  __shared__ float skv[NH * CH * CH];   // 12800 f
  __shared__ float sq[64 * C];          // 20480 f
  const int b = blockIdx.y;
  const int n0 = blockIdx.x * 64;
  const int cnt = min(64, N - n0);

  for (int e = threadIdx.x; e < NH * CH * CH; e += 256)
    skv[e] = kvb[(size_t)b * NH * CH * CH + e];
  for (int e = threadIdx.x; e < cnt * C; e += 256) {
    const int t = e / C, c = e % C;
    sq[t * C + c] = qkv[((size_t)b * N + n0 + t) * 960 + c];
  }
  __syncthreads();

  const float scale = 0.15811388300841898f;  // 40^-0.5
  for (int o = threadIdx.x; o < C * 64; o += 256) {
    const int nl = o % 64;
    const int c = o / 64;
    if (nl >= cnt) continue;
    const int h = c / CH, cv = c % CH;
    const float* qrow = &sq[nl * C + h * CH];
    const float* kvc = &skv[h * CH * CH + cv];
    float acc = 0.f;
    #pragma unroll
    for (int k = 0; k < CH; ++k) acc += qrow[k] * kvc[k * CH];
    acc *= scale;
    const int n = n0 + nl;
    if (n > 0) {
      const float cvv = conv_v[((size_t)b * C + c) * (HH * WW) + (n - 1)];
      acc += sq[nl * C + c] * cvv;
    }
    attn[((size_t)b * N + n) * C + c] = (_Float16)acc;
  }
}

// ---------------------------------------------------------------------------
extern "C" void kernel_launch(void* const* d_in, const int* in_sizes, int n_in,
                              void* d_out, int out_size, void* d_ws, size_t ws_size,
                              hipStream_t stream)
{
  using namespace coat;
  (void)in_sizes; (void)n_in; (void)out_size; (void)ws_size;
  const float* x     = (const float*)d_in[0];
  const float* Wqkv  = (const float*)d_in[1];
  const float* Wproj = (const float*)d_in[2];
  const float* bproj = (const float*)d_in[3];
  const float* cw3   = (const float*)d_in[6];
  const float* cb3   = (const float*)d_in[7];
  const float* cw5   = (const float*)d_in[8];
  const float* cb5   = (const float*)d_in[9];
  const float* cw7   = (const float*)d_in[10];
  const float* cb7   = (const float*)d_in[11];
  float* out = (float*)d_out;

  char* ws = (char*)d_ws;
  size_t off = 0;
  float* qkv = (float*)(ws + off);            off += (size_t)MTOT * 960 * sizeof(float);
  float* stats = (float*)(ws + off);          off += (size_t)B * C * sizeof(float);
  float* kvb = (float*)(ws + off);            off += (size_t)B * NH * CH * CH * sizeof(float);
  float* conv_v = (float*)(ws + off);         off += (size_t)B * C * HH * WW * sizeof(float);
  _Float16* attn = (_Float16*)(ws + off);     off += (size_t)MTOT * C * sizeof(_Float16);

  // stats + kvb are contiguous: one async memset clears both (ws is poisoned).
  hipMemsetAsync(stats, 0, (size_t)(B * C + B * NH * CH * CH) * sizeof(float), stream);

  // 1) qkv = x @ Wqkv
  {
    dim3 grid((MTOT + 127) / 128, 960 / 64);
    gemm_wmma_async<float, false><<<grid, 256, 0, stream>>>(
        x, C, Wqkv, 960, qkv, 960, nullptr, MTOT, 960, C);
  }
  // 2) softmax denominators
  k_sumexp<<<dim3(32, B), 320, 0, stream>>>(qkv, stats);
  // 3) kv
  k_kv<<<dim3(16, B * NH), 256, 0, stream>>>(qkv, stats, kvb);
  // 4) depthwise conv
  k_dwconv<<<dim3(7, C / 8, B), 256, 0, stream>>>(
      qkv, cw3, cb3, cw5, cb5, cw7, cb7, conv_v);
  // 5) combine -> f16 attn
  k_combine<<<dim3((N + 63) / 64, B), 256, 0, stream>>>(qkv, kvb, conv_v, attn);
  // 6) out = attn @ Wproj + bproj
  {
    dim3 grid((MTOT + 127) / 128, C / 64);
    gemm_wmma_async<_Float16, true><<<grid, 256, 0, stream>>>(
        attn, C, Wproj, C, out, C, bproj, MTOT, C, C);
  }
}
